// TransformerEncoderLayer_58171037057475
// MI455X (gfx1250) — matrix operand order repaired and checked by reading, not compiled, not run on hardware
//
#include <hip/hip_runtime.h>
#include <hip/hip_bf16.h>

// ---------------------------------------------------------------------------
// CDNA5 (gfx1250) transformer encoder layer.
//   D=256, H=8, DK=32, DFF=1024, B=2, S=4096.  wave32, WMMA f32_16x16x32_f16.
// All GEMMs run on v_wmma_f32_16x16x32_f16 (f16 A/B, f32 accumulate).
// Shared operand panels are double-buffered in LDS via the gfx1250 async
// global->LDS path (ASYNCcnt) when the toolchain exposes it.
// Attention is flash-style: never materializes the 4096x4096 score matrix.
// ---------------------------------------------------------------------------

#define S_LEN 4096
#define D_MODEL 256
#define N_HEADS 8
#define DK_HEAD 32
#define DFF_DIM 1024
#define BATCH 2
#define ROWS_TOTAL (BATCH * S_LEN)          // 8192
#define QK_SCALE 0.17677669529663689f       // 1/sqrt(32)

typedef __attribute__((ext_vector_type(16))) _Float16 v16h;
typedef __attribute__((ext_vector_type(8)))  float    v8f;
typedef int v4i __attribute__((vector_size(16)));

union HFrag { v16h f; uint4 u[2]; };

// ---- gfx1250 async global->LDS copy (ASYNCcnt) -----------------------------
#if defined(__has_builtin)
#if __has_builtin(__builtin_amdgcn_global_load_async_to_lds_b128) && \
    __has_builtin(__builtin_amdgcn_s_wait_asynccnt)
#define ASYNC_LDS 1
#endif
#endif
#ifndef ASYNC_LDS
#define ASYNC_LDS 0
#endif

typedef __attribute__((address_space(1))) v4i as1_v4i;
typedef __attribute__((address_space(3))) v4i as3_v4i;

__device__ __forceinline__ void cp16_async(void* lds_dst, const void* gsrc) {
#if ASYNC_LDS
    __builtin_amdgcn_global_load_async_to_lds_b128(
        (as1_v4i*)gsrc, (as3_v4i*)lds_dst, 0, 0);
#else
    *(uint4*)lds_dst = *(const uint4*)gsrc;
#endif
}
__device__ __forceinline__ void async_wait_all() {
#if ASYNC_LDS
    __builtin_amdgcn_s_wait_asynccnt(0);
#endif
}

// ---- fragment loaders (layouts per CDNA5 ISA 7.12.2, 16-bit wave32) --------
// A (16x32, row-major source, lda in elements):
//   lanes 0-15  : row m=lane, K = 0..7  (v0-3) and 16..23 (v4-7)
//   lanes 16-31 : row m=lane-16, K = 8..15 and 24..31
__device__ inline v16h load_a_frag(const _Float16* tile, int lda) {
    int lane = threadIdx.x & 31;
    int m = lane & 15, kh = lane >> 4;
    const _Float16* p = tile + (size_t)m * lda + kh * 8;
    HFrag r;
    r.u[0] = *(const uint4*)(p);
    r.u[1] = *(const uint4*)(p + 16);
    return r.f;
}

// B (32x16). Source is B^T stored N-major: bt[n][k], ldb in elements.
//   lanes 0-15  : col n=lane,   K = 0..7 and 16..23
//   lanes 16-31 : col n=lane-16,K = 8..15 and 24..31
__device__ inline v16h load_b_frag(const _Float16* bt, int ldb) {
    int lane = threadIdx.x & 31;
    int n = lane & 15, kh = lane >> 4;
    const _Float16* p = bt + (size_t)n * ldb + kh * 8;
    HFrag r;
    r.u[0] = *(const uint4*)(p);
    r.u[1] = *(const uint4*)(p + 16);
    return r.f;
}

__device__ inline v8f wmma_f16(v16h a, v16h b, v8f c) {
    return __builtin_amdgcn_wmma_f32_16x16x32_f16(false, a, false, b, (short)0, c,
                                                  false, false);
}

// C/D fragment element (j, lane): row = j + 8*(lane>>4), col = lane&15.

// ---------------------------------------------------------------------------
// Kernel: float -> f16 flat convert
__global__ void cvt_f32_to_f16(const float* __restrict__ in,
                               _Float16* __restrict__ out, long n) {
    long i = (long)blockIdx.x * blockDim.x + threadIdx.x;
    if (i < n) out[i] = (_Float16)in[i];
}

// Kernel: batched transpose + convert: in f32 [bt][R][C] -> out f16 [bt][C][R]
__global__ void transpose_cvt(const float* __restrict__ in,
                              _Float16* __restrict__ out,
                              int R, int C, long total) {
    long i = (long)blockIdx.x * blockDim.x + threadIdx.x;
    if (i >= total) return;
    long r = i % R;
    long rem = i / R;
    long c = rem % C;
    long bt = rem / C;
    out[i] = (_Float16)in[(bt * R + r) * C + c];
}

// ---------------------------------------------------------------------------
// Kernel 1: fused QKV projection.
//   grid = (B*H, S/128), block = 256 (8 waves). The head's full weight set
//   (3 x 32x256 f16 = 48KB) is async-staged into LDS once, then the K-loop is
//   entirely LDS-fed. Q gets 1/sqrt(DK) folded in.
//   Q,K stored f16 row-major [bh][S][32]; V stored transposed [bh][32][S].
__global__ __launch_bounds__(256) void qkv_kernel(
    const _Float16* __restrict__ srcH,   // [B][S][256] f16
    const _Float16* __restrict__ WqT,    // [H][32][256] f16
    const _Float16* __restrict__ WkT,
    const _Float16* __restrict__ WvT,
    const float* __restrict__ bq, const float* __restrict__ bk,
    const float* __restrict__ bv,
    _Float16* __restrict__ Q, _Float16* __restrict__ K,
    _Float16* __restrict__ VT) {
    __shared__ __align__(16) _Float16 Wsh[3][DK_HEAD][D_MODEL];

    int tid = threadIdx.x;
    int w = tid >> 5, lane = tid & 31;
    int bh = blockIdx.x, b = bh >> 3, h = bh & 7;
    int row0 = blockIdx.y * 128 + w * 16;

    const _Float16* Ab  = srcH + ((size_t)b * S_LEN + row0) * D_MODEL;
    const _Float16* WqB = WqT + (size_t)h * DK_HEAD * D_MODEL;
    const _Float16* WkB = WkT + (size_t)h * DK_HEAD * D_MODEL;
    const _Float16* WvB = WvT + (size_t)h * DK_HEAD * D_MODEL;

    // stage the head's Wq^T/Wk^T/Wv^T into LDS (each 1024 x 16B chunks)
    #pragma unroll
    for (int i = 0; i < 4; ++i) {
        int c = tid + 256 * i;            // 0..1023
        int row = c >> 5, seg = c & 31;   // row<32, seg*8 elem offset
        size_t o = (size_t)row * D_MODEL + seg * 8;
        cp16_async(&Wsh[0][row][seg * 8], WqB + o);
        cp16_async(&Wsh[1][row][seg * 8], WkB + o);
        cp16_async(&Wsh[2][row][seg * 8], WvB + o);
    }
    async_wait_all();
    __syncthreads();

    v8f aq[2], ak[2], av[2];
    #pragma unroll
    for (int t = 0; t < 2; ++t) { aq[t] = v8f{}; ak[t] = v8f{}; av[t] = v8f{}; }

    for (int kk = 0; kk < D_MODEL; kk += 32) {
        v16h a = load_a_frag(Ab + kk, D_MODEL);
        #pragma unroll
        for (int t = 0; t < 2; ++t) {
            aq[t] = wmma_f16(a, load_b_frag(&Wsh[0][16 * t][kk], D_MODEL), aq[t]);
            ak[t] = wmma_f16(a, load_b_frag(&Wsh[1][16 * t][kk], D_MODEL), ak[t]);
            av[t] = wmma_f16(a, load_b_frag(&Wsh[2][16 * t][kk], D_MODEL), av[t]);
        }
    }

    int n = lane & 15, mh = (lane >> 4) * 8;
    #pragma unroll
    for (int t = 0; t < 2; ++t) {
        int colk = 16 * t + n;
        float bqv = bq[h * DK_HEAD + colk];
        float bkv = bk[h * DK_HEAD + colk];
        float bvv = bv[h * DK_HEAD + colk];
        #pragma unroll
        for (int j = 0; j < 8; ++j) {
            int row = row0 + mh + j;
            size_t rowbase = ((size_t)bh * S_LEN + row) * DK_HEAD;
            Q[rowbase + colk] = (_Float16)((aq[t][j] + bqv) * QK_SCALE);
            K[rowbase + colk] = (_Float16)(ak[t][j] + bkv);
            VT[(size_t)bh * DK_HEAD * S_LEN + (size_t)colk * S_LEN + row] =
                (_Float16)(av[t][j] + bvv);
        }
    }
}

// ---------------------------------------------------------------------------
// Kernel 2: flash attention. grid = (B*H, S/128), block = 256 (8 waves).
// All waves share the block's 32-key K/V chunk, double-buffered in LDS via
// async copies. Each wave owns one 16-row query tile; 4 WMMAs per chunk.
__global__ __launch_bounds__(256) void flash_kernel(
    const _Float16* __restrict__ Q,     // [bh][S][32] (pre-scaled)
    const _Float16* __restrict__ K,     // [bh][S][32]
    const _Float16* __restrict__ VT,    // [bh][32][S]
    _Float16* __restrict__ ctx) {       // [B][S][H*32]
    __shared__ __align__(16) _Float16 Ksh[2][32][32];
    __shared__ __align__(16) _Float16 Vsh[2][32][32];
    __shared__ float sc[8][16][32];
    __shared__ __align__(16) _Float16 pt[8][16][32];
    __shared__ float sal[8][16];

    int tid = threadIdx.x;
    int w = tid >> 5, lane = tid & 31;
    int bh = blockIdx.x, b = bh >> 3, h = bh & 7;
    int row0 = blockIdx.y * 128 + w * 16;
    int n = lane & 15, mh = (lane >> 4) * 8;

    const _Float16* Qb = Q + ((size_t)bh * S_LEN + row0) * DK_HEAD;
    const _Float16* Kp = K + (size_t)bh * S_LEN * DK_HEAD;
    const _Float16* Vp = VT + (size_t)bh * DK_HEAD * S_LEN;

    auto stage = [&](int key0, int buf) {
        if (tid < 128) {                       // K chunk: 32 keys x 32 dk
            int r = tid >> 2, c8 = (tid & 3) * 8;
            cp16_async(&Ksh[buf][r][c8], Kp + (size_t)(key0 + r) * DK_HEAD + c8);
        } else {                               // V^T chunk: 32 dk x 32 keys
            int u = tid - 128;
            int r = u >> 2, c8 = (u & 3) * 8;
            cp16_async(&Vsh[buf][r][c8], Vp + (size_t)r * S_LEN + key0 + c8);
        }
    };

    v16h qa = load_a_frag(Qb, DK_HEAD);
    v8f c0 = v8f{}, c1 = v8f{};
    float mrow = -1e30f, lrow = 0.0f;

    stage(0, 0);
    const int nchunk = S_LEN / 32;
    for (int i = 0; i < nchunk; ++i) {
        int cur = i & 1;
        async_wait_all();
        __syncthreads();                       // chunk i resident; chunk i-1 reads done
        if (i + 1 < nchunk) stage((i + 1) * 32, cur ^ 1);

        v16h bk0 = load_b_frag(&Ksh[cur][0][0], 32);
        v16h bk1 = load_b_frag(&Ksh[cur][16][0], 32);
        v8f z = v8f{};
        v8f s0 = wmma_f16(qa, bk0, z);
        v8f s1 = wmma_f16(qa, bk1, z);

        #pragma unroll
        for (int j = 0; j < 8; ++j) {
            sc[w][mh + j][n]      = s0[j];
            sc[w][mh + j][n + 16] = s1[j];
        }
        __builtin_amdgcn_wave_barrier();       // wave-synchronous LDS handoff

        if (lane < 16) {
            float mc = mrow;
            #pragma unroll
            for (int c = 0; c < 32; ++c) mc = fmaxf(mc, sc[w][lane][c]);
            float al = __expf(mrow - mc);
            float sum = 0.0f;
            #pragma unroll
            for (int c = 0; c < 32; ++c) {
                float p = __expf(sc[w][lane][c] - mc);
                pt[w][lane][c] = (_Float16)p;
                sum += p;
            }
            lrow = lrow * al + sum;
            mrow = mc;
            sal[w][lane] = al;
        }
        __builtin_amdgcn_wave_barrier();

        #pragma unroll
        for (int j = 0; j < 8; ++j) {
            float a = sal[w][mh + j];
            c0[j] *= a;
            c1[j] *= a;
        }
        v16h pa  = load_a_frag(&pt[w][0][0], 32);
        v16h bv0 = load_b_frag(&Vsh[cur][0][0], 32);
        v16h bv1 = load_b_frag(&Vsh[cur][16][0], 32);
        c0 = wmma_f16(pa, bv0, c0);
        c1 = wmma_f16(pa, bv1, c1);
        __builtin_amdgcn_wave_barrier();
    }

    if (lane < 16) sal[w][lane] = 1.0f / lrow;
    __builtin_amdgcn_wave_barrier();

    _Float16* cb = ctx + ((size_t)b * S_LEN + row0) * D_MODEL + h * DK_HEAD;
    #pragma unroll
    for (int j = 0; j < 8; ++j) {
        float inv = sal[w][mh + j];
        cb[(size_t)(mh + j) * D_MODEL + n]      = (_Float16)(c0[j] * inv);
        cb[(size_t)(mh + j) * D_MODEL + 16 + n] = (_Float16)(c1[j] * inv);
    }
}

// ---------------------------------------------------------------------------
// Generic GEMM: out[row, col0+0..255] = A[row,:K] @ B[:K, col] + bias[col]
//   A: f16 row-major (lda), BT: f16 N-major [col][K].
//   The 256x32 B panel of each k-step (16KB) is async-staged into LDS,
//   double-buffered, shared by all 8 waves.
//   grid = (rows/128, Ntotal/256), block = 256 (wave = 16 rows x 256 cols)
//   mode: 0 = store f32, 1 = store f16, 2 = store f16 with ReLU
__global__ __launch_bounds__(256) void gemm_kernel(
    const _Float16* __restrict__ A, int lda,
    const _Float16* __restrict__ BT, int Kdim,
    const float* __restrict__ bias,
    void* __restrict__ out, int ldout, int mode) {
    __shared__ __align__(16) _Float16 Bsh[2][256][32];

    int tid = threadIdx.x;
    int w = tid >> 5, lane = tid & 31;
    int row0 = blockIdx.x * 128 + w * 16;
    int col0 = blockIdx.y * 256;

    auto stage = [&](int kk, int buf) {
        const _Float16* srcp = BT + (size_t)(col0 + tid) * Kdim + kk;
        #pragma unroll
        for (int c = 0; c < 4; ++c)
            cp16_async(&Bsh[buf][tid][c * 8], srcp + c * 8);
    };

    v8f acc[16];
    #pragma unroll
    for (int t = 0; t < 16; ++t) acc[t] = v8f{};

    stage(0, 0);
    const int nsteps = Kdim / 32;
    for (int s = 0; s < nsteps; ++s) {
        int cur = s & 1, kk = s * 32;
        async_wait_all();
        __syncthreads();
        if (s + 1 < nsteps) stage((s + 1) * 32, cur ^ 1);

        v16h a = load_a_frag(A + (size_t)row0 * lda + kk, lda);
        #pragma unroll
        for (int t = 0; t < 16; ++t)
            acc[t] = wmma_f16(a, load_b_frag(&Bsh[cur][16 * t][0], 32), acc[t]);
    }

    int n = lane & 15, mh = (lane >> 4) * 8;
    #pragma unroll
    for (int t = 0; t < 16; ++t) {
        int col = col0 + 16 * t + n;
        float bv = bias ? bias[col] : 0.0f;
        #pragma unroll
        for (int j = 0; j < 8; ++j) {
            int row = row0 + mh + j;
            float v = acc[t][j] + bv;
            if (mode == 2) v = fmaxf(v, 0.0f);
            if (mode == 0)
                ((float*)out)[(size_t)row * ldout + col] = v;
            else
                ((_Float16*)out)[(size_t)row * ldout + col] = (_Float16)v;
        }
    }
}

// ---------------------------------------------------------------------------
// LayerNorm over D=256. grid = rows, block = 256. z = x (+ res); out = LN(z).
__global__ __launch_bounds__(256) void ln_kernel(
    const float* __restrict__ x, const float* __restrict__ res,
    const float* __restrict__ g, const float* __restrict__ bb,
    float* __restrict__ outf, _Float16* __restrict__ outh) {
    __shared__ float s1[256];
    __shared__ float s2[256];
    int row = blockIdx.x, t = threadIdx.x;
    size_t idx = (size_t)row * D_MODEL + t;
    float v = x[idx];
    if (res) v += res[idx];
    s1[t] = v;
    s2[t] = v * v;
    __syncthreads();
    for (int off = 128; off > 0; off >>= 1) {
        if (t < off) { s1[t] += s1[t + off]; s2[t] += s2[t + off]; }
        __syncthreads();
    }
    float mu  = s1[0] * (1.0f / D_MODEL);
    float var = s2[0] * (1.0f / D_MODEL) - mu * mu;
    float rs  = rsqrtf(var + 1e-5f);
    float o = (v - mu) * rs * g[t] + bb[t];
    if (outf) outf[idx] = o;
    if (outh) outh[idx] = (_Float16)o;
}

// ---------------------------------------------------------------------------
extern "C" void kernel_launch(void* const* d_in, const int* in_sizes, int n_in,
                              void* d_out, int out_size, void* d_ws, size_t ws_size,
                              hipStream_t stream) {
    (void)in_sizes; (void)n_in; (void)out_size; (void)ws_size;

    const float* src   = (const float*)d_in[0];
    const float* Wq    = (const float*)d_in[1];
    const float* bq    = (const float*)d_in[2];
    const float* Wk    = (const float*)d_in[3];
    const float* bk    = (const float*)d_in[4];
    const float* Wv    = (const float*)d_in[5];
    const float* bv    = (const float*)d_in[6];
    const float* Wo    = (const float*)d_in[7];
    const float* bo    = (const float*)d_in[8];
    const float* ln1_g = (const float*)d_in[9];
    const float* ln1_b = (const float*)d_in[10];
    const float* W1    = (const float*)d_in[11];
    const float* b1    = (const float*)d_in[12];
    const float* W2    = (const float*)d_in[13];
    const float* b2    = (const float*)d_in[14];
    const float* ln2_g = (const float*)d_in[15];
    const float* ln2_b = (const float*)d_in[16];

    // workspace layout (bytes), 256-aligned slices; total ~66 MB
    char* ws = (char*)d_ws;
    size_t off = 0;
    auto take = [&](size_t bytes) { char* p = ws + off;
                                    off = (off + bytes + 255) & ~(size_t)255;
                                    return p; };
    _Float16* srcH = (_Float16*)take((size_t)ROWS_TOTAL * D_MODEL * 2);
    _Float16* WqT  = (_Float16*)take((size_t)N_HEADS * DK_HEAD * D_MODEL * 2);
    _Float16* WkT  = (_Float16*)take((size_t)N_HEADS * DK_HEAD * D_MODEL * 2);
    _Float16* WvT  = (_Float16*)take((size_t)N_HEADS * DK_HEAD * D_MODEL * 2);
    _Float16* WoT  = (_Float16*)take((size_t)D_MODEL * D_MODEL * 2);
    _Float16* W1T  = (_Float16*)take((size_t)DFF_DIM * D_MODEL * 2);
    _Float16* W2T  = (_Float16*)take((size_t)D_MODEL * DFF_DIM * 2);
    _Float16* Qb   = (_Float16*)take((size_t)BATCH * N_HEADS * S_LEN * DK_HEAD * 2);
    _Float16* Kb   = (_Float16*)take((size_t)BATCH * N_HEADS * S_LEN * DK_HEAD * 2);
    _Float16* VTb  = (_Float16*)take((size_t)BATCH * N_HEADS * DK_HEAD * S_LEN * 2);
    _Float16* ctx  = (_Float16*)take((size_t)ROWS_TOTAL * D_MODEL * 2);
    float*    y    = (float*)   take((size_t)ROWS_TOTAL * D_MODEL * 4);
    _Float16* x1h  = (_Float16*)take((size_t)ROWS_TOTAL * D_MODEL * 2);
    float*    x1f  = (float*)   take((size_t)ROWS_TOTAL * D_MODEL * 4);
    _Float16* hmid = (_Float16*)take((size_t)ROWS_TOTAL * DFF_DIM * 2);
    float*    y2   = (float*)   take((size_t)ROWS_TOTAL * D_MODEL * 4);

    // 0) convert src to f16 and transpose/convert all weights to N-major f16
    {
        long n = (long)ROWS_TOTAL * D_MODEL;
        cvt_f32_to_f16<<<dim3((n + 255) / 256), dim3(256), 0, stream>>>(src, srcH, n);
        long tq = (long)N_HEADS * D_MODEL * DK_HEAD;
        transpose_cvt<<<dim3((tq + 255) / 256), dim3(256), 0, stream>>>(Wq, WqT, D_MODEL, DK_HEAD, tq);
        transpose_cvt<<<dim3((tq + 255) / 256), dim3(256), 0, stream>>>(Wk, WkT, D_MODEL, DK_HEAD, tq);
        transpose_cvt<<<dim3((tq + 255) / 256), dim3(256), 0, stream>>>(Wv, WvT, D_MODEL, DK_HEAD, tq);
        long to = (long)D_MODEL * D_MODEL;
        transpose_cvt<<<dim3((to + 255) / 256), dim3(256), 0, stream>>>(Wo, WoT, D_MODEL, D_MODEL, to);
        long t1 = (long)D_MODEL * DFF_DIM;
        transpose_cvt<<<dim3((t1 + 255) / 256), dim3(256), 0, stream>>>(W1, W1T, D_MODEL, DFF_DIM, t1);
        transpose_cvt<<<dim3((t1 + 255) / 256), dim3(256), 0, stream>>>(W2, W2T, DFF_DIM, D_MODEL, t1);
    }

    // 1) QKV projection  (grid: B*H x S/128)
    qkv_kernel<<<dim3(BATCH * N_HEADS, S_LEN / 128), dim3(256), 0, stream>>>(
        srcH, WqT, WkT, WvT, bq, bk, bv, Qb, Kb, VTb);

    // 2) flash attention -> ctx (heads concatenated)
    flash_kernel<<<dim3(BATCH * N_HEADS, S_LEN / 128), dim3(256), 0, stream>>>(
        Qb, Kb, VTb, ctx);

    // 3) out projection: y = ctx @ Wo + bo  (f32)
    gemm_kernel<<<dim3(ROWS_TOTAL / 128, 1), dim3(256), 0, stream>>>(
        ctx, D_MODEL, WoT, D_MODEL, bo, (void*)y, D_MODEL, 0);

    // LN1: x1 = LN(src + y)
    ln_kernel<<<dim3(ROWS_TOTAL), dim3(256), 0, stream>>>(
        y, src, ln1_g, ln1_b, x1f, x1h);

    // 4) FFN up: hmid = relu(x1 @ W1 + b1)  (f16)
    gemm_kernel<<<dim3(ROWS_TOTAL / 128, DFF_DIM / 256), dim3(256), 0, stream>>>(
        x1h, D_MODEL, W1T, D_MODEL, b1, (void*)hmid, DFF_DIM, 2);

    // 5) FFN down: y2 = hmid @ W2 + b2  (f32)
    gemm_kernel<<<dim3(ROWS_TOTAL / 128, 1), dim3(256), 0, stream>>>(
        hmid, DFF_DIM, W2T, DFF_DIM, b2, (void*)y2, D_MODEL, 0);

    // LN2: out = LN(x1 + y2)
    ln_kernel<<<dim3(ROWS_TOTAL), dim3(256), 0, stream>>>(
        y2, x1f, ln2_g, ln2_b, (float*)d_out, nullptr);
}
